// BertUnpadSelfAttention_46548855554786
// MI455X (gfx1250) — matrix operand order repaired and checked by reading, not compiled, not run on hardware
//
#include <hip/hip_runtime.h>
#include <hip/hip_bf16.h>
#include <stdint.h>

// ---------------------------------------------------------------------------
// BertUnpadSelfAttention for MI455X (gfx1250, wave32, WMMA).
//   B=4 S=1024 L=512 H=12 D=64 DIM=768  -> bandwidth-bound on bias (96MB, NT).
//   bf16 WMMA (v_wmma_f32_16x16x32_bf16) with f32 accumulation everywhere.
//   One-shot f32->bf16 convert pass so GEMM/attention hot loops are pure
//   global_load_b128 + v_wmma (no per-iteration convert VALU).
// ---------------------------------------------------------------------------

typedef __attribute__((ext_vector_type(16))) __bf16 v16bf;
typedef __attribute__((ext_vector_type(8)))  float  v8f;

#define B_    4
#define S_    1024
#define L_    512
#define H_    12
#define D_    64
#define DIM_  768
#define NQKV_ 2304
#define NNZ_  2048
#define BUF_ELEMS (B_*H_*S_*D_)      // 3,145,728 bf16 per Q/K/V buffer (6 MB)

// workspace layout (bytes)
#define QKV_BYTES   (3u * BUF_ELEMS * 2u)              // 18,874,368
#define HBF_OFF_E   (3u * BUF_ELEMS)                   // ushort offset of hidden-bf16
#define WBF_OFF_E   (3u * BUF_ELEMS + NNZ_ * DIM_)     // ushort offset of W-bf16

static __device__ __forceinline__ unsigned short f2bfbits(float f) {
  unsigned u = __builtin_bit_cast(unsigned, f);
  unsigned r = u + 0x7FFFu + ((u >> 16) & 1u);   // round-to-nearest-even
  return (unsigned short)(r >> 16);
}
static __device__ __forceinline__ __bf16 bfbits(unsigned short v) {
  return __builtin_bit_cast(__bf16, v);
}
static __device__ __forceinline__ __bf16 f2bf(float f) { return bfbits(f2bfbits(f)); }

// ---------------------------------------------------------------------------
// Kernel 0: zero the padded bf16 Q/K/V workspace (padded keys must be 0 so
// softmax over k>=L sees score==bias and PV contribution 0, as in reference).
// ---------------------------------------------------------------------------
__global__ __launch_bounds__(256) void zero_ws_kernel(float4* __restrict__ ws, int n16) {
  int i = blockIdx.x * 256 + threadIdx.x;
  if (i < n16) ws[i] = make_float4(0.f, 0.f, 0.f, 0.f);
}

// ---------------------------------------------------------------------------
// Kernel 1: one-shot f32 -> bf16 (RNE) pack, 4 elements/thread.
// ---------------------------------------------------------------------------
__global__ __launch_bounds__(256) void cvt_f32_bf16_kernel(
    const float4* __restrict__ src, ushort4* __restrict__ dst, int n4) {
  int i = blockIdx.x * 256 + threadIdx.x;
  if (i < n4) {
    float4 v = src[i];
    ushort4 o;
    o.x = f2bfbits(v.x); o.y = f2bfbits(v.y);
    o.z = f2bfbits(v.z); o.w = f2bfbits(v.w);
    dst[i] = o;
  }
}

// ---------------------------------------------------------------------------
// Kernel 2: QKV projection  qkv[m,n] = sum_d hidden[m,d]*W[n,d] + b[n],
// scattered at indices[m] into padded bf16 buffers:
//   Q,K : [B][H][S][D]      V : [B][H][D][S]  (transposed for PV B-fragments)
// One wave computes a 16(M) x 64(N) tile; A-fragment reused across 4 WMMAs.
// ---------------------------------------------------------------------------
__global__ __launch_bounds__(256) void qkv_gemm_kernel(
    const unsigned short* __restrict__ hbf, const unsigned short* __restrict__ wbf,
    const float* __restrict__ Wb, const int* __restrict__ indices,
    unsigned short* __restrict__ qkvbuf) {
  const int lane = threadIdx.x & 31;
  const int wg   = blockIdx.x * 8 + (threadIdx.x >> 5);
  const int NT   = NQKV_ / 64;              // 36 n-supertiles
  const int mt   = wg / NT, nt = wg % NT;
  const int m0   = mt * 16, n0 = nt * 64;
  const int half = (lane < 16) ? 0 : 1;
  const int lx   = lane & 15;

  v8f acc[4];
  #pragma unroll
  for (int t = 0; t < 4; ++t) acc[t] = (v8f){};

  const int arow = m0 + lx;
  for (int c = 0; c < DIM_ / 32; ++c) {       // 24 K-steps of 32
    const int base = c * 32;
    // A fragment (16x32 bf16): lane<16 holds K 0-15; lane>=16 K 16-31
    // -> one contiguous 16xbf16 (32B) run per lane.
    v16bf a;
    {
      const unsigned short* hp = hbf + arow * DIM_ + base + half * 16;
      #pragma unroll
      for (int e = 0; e < 16; ++e) a[e] = bfbits(hp[e]);
    }
    #pragma unroll
    for (int t = 0; t < 4; ++t) {
      // B fragment (32x16): col n = lane&15; K rows e(+16 for hi half);
      // B[d,n] = W[n, base+d] -> 16 contiguous bf16 (32B) per lane.
      const int ncol = n0 + t * 16 + lx;
      const unsigned short* wp = wbf + ncol * DIM_ + base + half * 16;
      v16bf b;
      #pragma unroll
      for (int e = 0; e < 16; ++e) b[e] = bfbits(wp[e]);
      acc[t] = __builtin_amdgcn_wmma_f32_16x16x32_bf16(
          false, a, false, b, (short)0, acc[t], false, false);
    }
  }

  // Epilogue: + bias, scatter through indices into padded bf16 buffers.
  #pragma unroll
  for (int t = 0; t < 4; ++t) {
    const int ncol  = n0 + t * 16 + lx;
    const float bv  = Wb[ncol];
    const int which = ncol / DIM_;            // 0=Q 1=K 2=V
    const int rem   = ncol % DIM_;
    const int h     = rem / D_, d = rem % D_;
    unsigned short* buf = qkvbuf + which * BUF_ELEMS;
    #pragma unroll
    for (int r = 0; r < 8; ++r) {             // C layout: row = r (+8 hi half)
      const int m  = m0 + r + half * 8;
      const int p  = indices[m];
      const int b_ = p / S_, s_ = p % S_;
      const unsigned short val = f2bfbits(acc[t][r] + bv);
      if (which == 2) buf[((b_ * H_ + h) * D_ + d) * S_ + s_] = val;   // V^T
      else            buf[((b_ * H_ + h) * S_ + s_) * D_ + d] = val;   // Q,K
    }
  }
}

// ---------------------------------------------------------------------------
// Kernel 3: flash attention. One wave per (b,h,16 q-rows); loop over 32-key
// tiles of the FULL S=1024 (padded keys have K=V=0, matching the reference's
// unmasked softmax). Bias streamed with non-temporal loads (no reuse).
// ---------------------------------------------------------------------------
__global__ __launch_bounds__(256) void attn_kernel(
    const unsigned short* __restrict__ qkvbuf,
    const float* __restrict__ bias,
    const int* __restrict__ cu_seqlens,
    float* __restrict__ out) {
  __shared__ float pshare[8 * 16 * 32];       // per-wave P-transpose buffer
  const int lane = threadIdx.x & 31;
  const int wv   = threadIdx.x >> 5;
  const int wg   = blockIdx.x * 8 + wv;
  const int QT   = L_ / 16;                   // 32 q-tiles per (b,h)
  const int qt   = wg % QT;
  const int h    = (wg / QT) % H_;
  const int b    = wg / (QT * H_);
  const int q0   = qt * 16;
  const int half = (lane < 16) ? 0 : 1;
  const int lx   = lane & 15;
  float* pb = pshare + wv * (16 * 32);

  const unsigned short* Q  = qkvbuf;                     // [B][H][S][D]
  const unsigned short* K  = qkvbuf + BUF_ELEMS;         // [B][H][S][D]
  const unsigned short* Vt = qkvbuf + 2 * BUF_ELEMS;     // [B][H][D][S]
  const int bh = b * H_ + h;

  // Q A-fragments for d 0..31 and 32..63 (reused across all key tiles).
  v16bf aq[2];
  {
    const unsigned short* qp = Q + (bh * S_ + q0 + lx) * D_;
    #pragma unroll
    for (int c = 0; c < 2; ++c) {
      const unsigned short* qpc = qp + c * 32 + half * 16;
      #pragma unroll
      for (int e = 0; e < 16; ++e) aq[c][e] = bfbits(qpc[e]);
    }
  }

  float mi[8], li[8];
  #pragma unroll
  for (int r = 0; r < 8; ++r) { mi[r] = -1e30f; li[r] = 0.f; }
  v8f acc[4];
  #pragma unroll
  for (int t = 0; t < 4; ++t) acc[t] = (v8f){};

  const float scale = 0.125f;                 // 1/sqrt(64)

  for (int kt = 0; kt < S_ / 32; ++kt) {
    const int k0 = kt * 32;
    float sc[2][8];
    // ---- scores: two 16q x 16k WMMA C-tiles (keys k0..+15, k0+16..+31) ----
    #pragma unroll
    for (int j = 0; j < 2; ++j) {
      const int kb = k0 + j * 16;
      v8f cs = (v8f){};
      #pragma unroll
      for (int c = 0; c < 2; ++c) {
        // B[d,n] = K[kb+n, 32c+d]; 16 contiguous bf16 (32B) per lane.
        const unsigned short* kp = K + (bh * S_ + kb + lx) * D_ + c * 32 + half * 16;
        v16bf bk;
        #pragma unroll
        for (int e = 0; e < 16; ++e) bk[e] = bfbits(kp[e]);
        cs = __builtin_amdgcn_wmma_f32_16x16x32_bf16(
            false, aq[c], false, bk, (short)0, cs, false, false);
      }
      #pragma unroll
      for (int r = 0; r < 8; ++r) {
        const int row = r + half * 8;
        const float bij = __builtin_nontemporal_load(
            bias + ((bh * S_) + q0 + row) * S_ + kb + lx);
        sc[j][r] = cs[r] * scale + bij;
      }
    }
    // ---- online softmax (row reductions inside 16-lane halves) ----
    float alpha[8];
    #pragma unroll
    for (int r = 0; r < 8; ++r) {
      float mx = fmaxf(sc[0][r], sc[1][r]);
      #pragma unroll
      for (int m = 1; m < 16; m <<= 1) mx = fmaxf(mx, __shfl_xor(mx, m, 32));
      const float mnew = fmaxf(mi[r], mx);
      alpha[r] = __expf(mi[r] - mnew);
      sc[0][r] = __expf(sc[0][r] - mnew);
      sc[1][r] = __expf(sc[1][r] - mnew);
      float sm = sc[0][r] + sc[1][r];
      #pragma unroll
      for (int m = 1; m < 16; m <<= 1) sm += __shfl_xor(sm, m, 32);
      li[r] = li[r] * alpha[r] + sm;
      mi[r] = mnew;
    }
    #pragma unroll
    for (int t = 0; t < 4; ++t)
      #pragma unroll
      for (int r = 0; r < 8; ++r) acc[t][r] *= alpha[r];

    // ---- transpose P (C layout -> A layout) through per-wave LDS ----
    #pragma unroll
    for (int j = 0; j < 2; ++j)
      #pragma unroll
      for (int r = 0; r < 8; ++r)
        pb[(r + half * 8) * 32 + j * 16 + lx] = sc[j][r];
    __asm__ volatile("s_wait_dscnt 0" ::: "memory");  // cross-lane LDS dep
    v16bf pa;
    #pragma unroll
    for (int e = 0; e < 16; ++e) {
      const int col = e + half * 16;
      pa[e] = f2bf(pb[lx * 32 + col]);
    }
    // ---- PV: 4 d-tiles of 16; B[k,n] = Vt[16t+n, k0+k] -> contiguous ----
    #pragma unroll
    for (int t = 0; t < 4; ++t) {
      const unsigned short* vp = Vt + (bh * D_ + t * 16 + lx) * S_ + k0 + half * 16;
      v16bf bv;
      #pragma unroll
      for (int e = 0; e < 16; ++e) bv[e] = bfbits(vp[e]);
      acc[t] = __builtin_amdgcn_wmma_f32_16x16x32_bf16(
          false, pa, false, bv, (short)0, acc[t], false, false);
    }
  }

  // ---- normalize and gather-store: out row = cu_seqlens[b] + q ----
  const int tok0 = cu_seqlens[b] + q0;
  #pragma unroll
  for (int t = 0; t < 4; ++t) {
    #pragma unroll
    for (int r = 0; r < 8; ++r) {
      const int row = r + half * 8;
      out[(tok0 + row) * (H_ * D_) + h * D_ + t * 16 + lx] = acc[t][r] / li[r];
    }
  }
}

// ---------------------------------------------------------------------------
extern "C" void kernel_launch(void* const* d_in, const int* in_sizes, int n_in,
                              void* d_out, int out_size, void* d_ws, size_t ws_size,
                              hipStream_t stream) {
  const float* hidden = (const float*)d_in[0];   // (2048, 768)
  const float* Wqkv_w = (const float*)d_in[1];   // (2304, 768)
  const float* Wqkv_b = (const float*)d_in[2];   // (2304,)
  const float* bias   = (const float*)d_in[3];   // (4, 12, 1024, 1024)
  const int*   indices    = (const int*)d_in[4]; // (2048,)
  const int*   cu_seqlens = (const int*)d_in[6]; // (5,)
  float* out = (float*)d_out;                    // (2048, 768)

  unsigned short* wsbase = (unsigned short*)d_ws;
  unsigned short* qkvbuf = wsbase;                 // 3 x 6MB bf16
  unsigned short* hbf    = wsbase + HBF_OFF_E;     // hidden bf16 (3 MB)
  unsigned short* wbf    = wsbase + WBF_OFF_E;     // W bf16 (3.5 MB)

  // 0) zero padded Q/K/V workspace (18 MB -> 1,179,648 x float4)
  {
    const int n16 = (int)(QKV_BYTES / 16);
    zero_ws_kernel<<<(n16 + 255) / 256, 256, 0, stream>>>((float4*)d_ws, n16);
  }
  // 1) one-shot f32->bf16 packs (hidden: 393216 x4, W: 442368 x4)
  {
    const int nh4 = (NNZ_ * DIM_) / 4;
    cvt_f32_bf16_kernel<<<(nh4 + 255) / 256, 256, 0, stream>>>(
        (const float4*)hidden, (ushort4*)hbf, nh4);
    const int nw4 = (NQKV_ * DIM_) / 4;
    cvt_f32_bf16_kernel<<<(nw4 + 255) / 256, 256, 0, stream>>>(
        (const float4*)Wqkv_w, (ushort4*)wbf, nw4);
  }
  // 2) QKV projection + scatter: 128 m-tiles x 36 n-supertiles = 4608 waves
  {
    const int waves = (NNZ_ / 16) * (NQKV_ / 64);
    qkv_gemm_kernel<<<waves / 8, 256, 0, stream>>>(hbf, wbf, Wqkv_b,
                                                   indices, qkvbuf);
  }
  // 3) flash attention + gather: 4*12*32 = 1536 waves
  {
    const int waves = B_ * H_ * (L_ / 16);
    attn_kernel<<<waves / 8, 256, 0, stream>>>(qkvbuf, bias, cu_seqlens, out);
  }
}
